// Linear4Bit_50216757625007
// MI455X (gfx1250) — compile-verified, matrix operand order checked by reading, not run on hardware
//
#include <hip/hip_runtime.h>

// 4-bit quantized GEMM: out[M,N] = x[M,K] * dequant(packed_weight[K/8,N], scales[K/128,N])
// dequant: w = (nibble - 8) / 7.5 * scale
// Compute-bound (AI ~2.7 kFLOP/B) -> V_WMMA_F32_16X16X32_F16, double-buffered LDS pipeline,
// packed fp16 bit-trick dequant (v_perm + v_pk_add_f16 + v_pk_mul_f16).

typedef __attribute__((ext_vector_type(16))) _Float16 v16h;
typedef __attribute__((ext_vector_type(8)))  _Float16 v8h;
typedef __attribute__((ext_vector_type(2)))  _Float16 v2h;
typedef __attribute__((ext_vector_type(8)))  float    v8f;

#define BM 128
#define BN 128
#define BK 64
#define LDP 72                    // padded row stride in halves (144B): conflict-free b128 reads
#define TILE_HALF (BM * LDP)

union FragAB {
    v16h v;
    struct { v8h lo; v8h hi; } p;
};

// Dequant one int32 (8 nibbles along K) -> 8 contiguous fp16 in LDS [n][k] layout.
// 0x6400|nib == fp16(1024+nib) exactly; (1024+nib)-1032 is exact (Sterbenz);
// only the final *sh rounds, matching the inherent f16 WMMA input rounding.
__device__ __forceinline__ void dequant_store(_Float16* dst, int p, _Float16 sh) {
    const unsigned e = (unsigned)p & 0x0F0F0F0Fu;          // even nibbles in bytes 0..3
    const unsigned o = ((unsigned)p >> 4) & 0x0F0F0F0Fu;   // odd  nibbles in bytes 0..3
    v2h s2;   s2[0] = sh; s2[1] = sh;
    v2h bias; bias[0] = (_Float16)(-1032.0f); bias[1] = (_Float16)(-1032.0f);
    v8h w;
    #pragma unroll
    for (int j = 0; j < 4; ++j) {
        // result bytes: [e.b(j), 0x00, o.b(j), 0x00]  (perm: sel 4..7 -> src0, 0..3 -> src1)
        const unsigned sel  = 0x0C000C04u + (unsigned)j * 0x00010001u;
        const unsigned bits = __builtin_amdgcn_perm(e, o, sel) | 0x64006400u;
        const v2h h2 = __builtin_bit_cast(v2h, bits);
        const v2h w2 = (h2 + bias) * s2;                   // v_pk_add_f16 + v_pk_mul_f16
        w[2 * j]     = w2[0];
        w[2 * j + 1] = w2[1];
    }
    *(v8h*)dst = w;
}

__device__ __forceinline__ void compute_tile(const _Float16* __restrict__ Acur,
                                             const _Float16* __restrict__ Bcur,
                                             int wm, int wn, int fm, int fk,
                                             v8f acc[2][4]) {
    #pragma unroll
    for (int ks = 0; ks < 2; ++ks) {
        const int kb = ks * 32;
        FragAB a[2], b[4];
        #pragma unroll
        for (int mt = 0; mt < 2; ++mt) {
            const int r = wm * 32 + mt * 16 + fm;
            a[mt].p.lo = *(const v8h*)&Acur[r * LDP + kb + fk];
            a[mt].p.hi = *(const v8h*)&Acur[r * LDP + kb + fk + 16];
        }
        #pragma unroll
        for (int nt = 0; nt < 4; ++nt) {
            const int c = wn * 64 + nt * 16 + fm;
            b[nt].p.lo = *(const v8h*)&Bcur[c * LDP + kb + fk];
            b[nt].p.hi = *(const v8h*)&Bcur[c * LDP + kb + fk + 16];
        }
        #pragma unroll
        for (int mt = 0; mt < 2; ++mt) {
            #pragma unroll
            for (int nt = 0; nt < 4; ++nt) {
                acc[mt][nt] = __builtin_amdgcn_wmma_f32_16x16x32_f16(
                    false, a[mt].v, false, b[nt].v,
                    (short)0, acc[mt][nt], false, false);
            }
        }
    }
}

__global__ __launch_bounds__(256)
void q4gemm_wmma_f16(const _Float16* __restrict__ x,
                     const int*      __restrict__ pw,
                     const _Float16* __restrict__ sc,
                     _Float16*       __restrict__ out,
                     int M, int N, int K)
{
    __shared__ _Float16 As[2 * TILE_HALF];   // ping-pong [m][k] tiles of x (128x64)
    __shared__ _Float16 Bs[2 * TILE_HALF];   // ping-pong [n][k] tiles of dequantized W

    const int tid  = threadIdx.x;            // 256 threads = 8 wave32
    const int lane = tid & 31;
    const int wave = tid >> 5;
    const int wm   = wave & 3;               // 4 wave-rows  -> 32 M each
    const int wn   = wave >> 2;              // 2 wave-cols  -> 64 N each

    const int rowBlock = blockIdx.y * BM;
    const int colBlock = blockIdx.x * BN;

    // A staging: 2 threads per row, 32 halves each (4x b128)
    const int arow  = tid >> 1;
    const int acolh = (tid & 1) << 5;

    // B staging: 8 packed rows x 128 cols = 1024 int32; 4 per thread, rows bk8+{0,2,4,6}
    const int bn  = tid & 127;
    const int bk8 = tid >> 7;

    const _Float16* aSrc = x  + (size_t)(rowBlock + arow) * K + acolh;
    const int*      bSrc = pw + (size_t)bk8 * N + (colBlock + bn);
    const _Float16* sSrc = sc + (colBlock + bn);

    v8f acc[2][4] = {};

    // WMMA 16-bit fragment addressing: lane<16 K-chunks {0..7},{16..23}; lane>=16 {8..15},{24..31}
    const int fm = lane & 15;
    const int fk = (lane >> 4) << 3;

    const int T = K / BK;                    // 64 tiles

    // ---------------- prologue: fetch + stage tile 0 into buffer 0 ----------------
    {
        v8h a0[4]; int b0[4];
        #pragma unroll
        for (int i = 0; i < 4; ++i) a0[i] = *(const v8h*)(aSrc + i * 8);
        #pragma unroll
        for (int i = 0; i < 4; ++i) b0[i] = bSrc[(size_t)(2 * i) * N];
        const _Float16 s0 = (_Float16)((float)sSrc[0] * (1.0f / 7.5f));

        #pragma unroll
        for (int i = 0; i < 4; ++i)
            *(v8h*)&As[arow * LDP + acolh + i * 8] = a0[i];
        #pragma unroll
        for (int i = 0; i < 4; ++i)
            dequant_store(&Bs[bn * LDP + ((bk8 + 2 * i) << 3)], b0[i], s0);
    }
    __syncthreads();

    // ---------------- steady state: unconditional fetch/compute/stage ----------------
    for (int t = 0; t < T - 1; ++t) {
        const int cur = t & 1;

        // fetch tile t+1 (loads overlap the WMMAs; waits sink to first use in staging)
        v8h anx[4]; int bnx[4];
        const _Float16* as2 = aSrc + (size_t)(t + 1) * BK;
        #pragma unroll
        for (int i = 0; i < 4; ++i) anx[i] = *(const v8h*)(as2 + i * 8);
        __builtin_prefetch(as2 + BK, 0, 1);                       // tile t+2 hint
        const int* bs2 = bSrc + (size_t)(t + 1) * (BK / 8) * N;
        #pragma unroll
        for (int i = 0; i < 4; ++i) bnx[i] = bs2[(size_t)(2 * i) * N];
        __builtin_prefetch(bs2 + (size_t)(BK / 8) * N, 0, 1);
        const _Float16 snx = (_Float16)((float)sSrc[(size_t)((t + 1) >> 1) * N] * (1.0f / 7.5f));

        // compute tile t from current buffer
        compute_tile(As + cur * TILE_HALF, Bs + cur * TILE_HALF, wm, wn, fm, fk, acc);

        // stage tile t+1 into the other buffer (its readers finished before last barrier)
        _Float16* Anxt = As + (cur ^ 1) * TILE_HALF;
        _Float16* Bnxt = Bs + (cur ^ 1) * TILE_HALF;
        #pragma unroll
        for (int i = 0; i < 4; ++i)
            *(v8h*)&Anxt[arow * LDP + acolh + i * 8] = anx[i];
        #pragma unroll
        for (int i = 0; i < 4; ++i)
            dequant_store(&Bnxt[bn * LDP + ((bk8 + 2 * i) << 3)], bnx[i], snx);
        __syncthreads();
    }

    // tail: compute last tile
    compute_tile(As + ((T - 1) & 1) * TILE_HALF, Bs + ((T - 1) & 1) * TILE_HALF,
                 wm, wn, fm, fk, acc);

    // ---------------- epilogue: C/D layout -> row = r + (lane>=16 ? 8 : 0), col = lane&15 ----
    const int colb = colBlock + wn * 64 + (lane & 15);
    const int rowb = rowBlock + wm * 32 + ((lane >> 4) << 3);
    #pragma unroll
    for (int mt = 0; mt < 2; ++mt) {
        #pragma unroll
        for (int nt = 0; nt < 4; ++nt) {
            #pragma unroll
            for (int r = 0; r < 8; ++r) {
                const int row = rowb + mt * 16 + r;
                const int col = colb + nt * 16;
                out[(size_t)row * N + col] = (_Float16)acc[mt][nt][r];
            }
        }
    }
}

extern "C" void kernel_launch(void* const* d_in, const int* in_sizes, int n_in,
                              void* d_out, int out_size, void* d_ws, size_t ws_size,
                              hipStream_t stream) {
    const _Float16* x   = (const _Float16*)d_in[0];  // [M, K] fp16
    const int*      pw  = (const int*)d_in[1];       // [K/8, N] int32
    const _Float16* sc  = (const _Float16*)d_in[2];  // [K/128, N] fp16
    _Float16*       out = (_Float16*)d_out;          // [M, N] fp16

    const int K = 4096;
    const int N = in_sizes[1] / (K / 8);             // 11008
    const int M = in_sizes[0] / K;                   // 8192

    dim3 grid(N / BN, M / BM);                       // 86 x 64, exact division
    q4gemm_wmma_f16<<<grid, 256, 0, stream>>>(x, pw, sc, out, M, N, K);
}
